// Encoder_3994319585450
// MI455X (gfx1250) — matrix operand order repaired and checked by reading, not compiled, not run on hardware
//
#include <hip/hip_runtime.h>
#include <stdint.h>

// ---------------------------------------------------------------------------
// LSTM encoder for MI455X (gfx1250): bf16 WMMA, weight-resident recurrence.
//   B=64, T=512, D=128, H=512, DS=32
// Phase 0: swizzle x / kernel / rec_kernel / dense_w into WMMA fragment layouts
// Phase 1: xw = x@kernel + bias (bf16 WMMA, stored bf16 in C-fragment layout)
// Phase 2: 32 workgroups (2 batch groups x 16 N-split WGs), 2 waves each.
//          Each wave holds its rec_kernel slice (64 fragments = 512 VGPRs) in
//          registers for all 512 steps; h exchanged via double-buffered L2
//          buffer + per-group spin barrier; c in VGPRs; dense fused.
// Workspace: xswz 8MB @0 | kswz .5MB @8MB | rswz 2MB @9MB | dswz 32KB @11MB |
//            xw 128MB @12MB | hbuf 128KB @140MB | ctr 4KB @140MB+128KB
// ---------------------------------------------------------------------------

typedef __attribute__((ext_vector_type(16))) __bf16 bf16x16;
typedef __attribute__((ext_vector_type(8)))  float  f32x8;

union FragB { uint4 q[2]; bf16x16 v; };

__device__ __forceinline__ uint32_t bf16rn(float f) {
  uint32_t u = __float_as_uint(f);
  return (u + 0x7FFFu + ((u >> 16) & 1u)) >> 16;   // round-to-nearest-even
}
__device__ __forceinline__ uint32_t pack2(float a, float b) {
  return bf16rn(a) | (bf16rn(b) << 16);
}
__device__ __forceinline__ float bflo(uint32_t u) { return __uint_as_float(u << 16); }
__device__ __forceinline__ float bfhi(uint32_t u) { return __uint_as_float(u & 0xFFFF0000u); }
__device__ __forceinline__ float sigmoidf_(float x) { return 1.0f / (1.0f + __expf(-x)); }

// ---- Phase 0a: x[B,T,D] -> A-fragment layout, row index r = t*64 + b -------
__global__ void swizzle_x_kernel(const float* __restrict__ x, uint32_t* __restrict__ dst) {
  const int total = 2048 * 4 * 32 * 8;
  for (int tid = blockIdx.x * blockDim.x + threadIdx.x; tid < total;
       tid += gridDim.x * blockDim.x) {
    int v = tid & 7, lane = (tid >> 3) & 31, ktile = (tid >> 8) & 3, mtile = tid >> 10;
    int half = lane >> 4;
    int klocal = (v < 4) ? (2 * v + 8 * half) : (16 + 2 * (v - 4) + 8 * half);
    int k = ktile * 32 + klocal;
    int r = mtile * 16 + (lane & 15);            // r = t*64 + b
    int t = r >> 6, b = r & 63;
    const float* p = x + ((size_t)b * 512 + t) * 128 + k;
    dst[tid] = pack2(p[0], p[1]);
  }
}

// ---- Phase 0b: weight [K,N] row-major -> B-fragment layout -----------------
__global__ void swizzle_w_kernel(const float* __restrict__ src, uint32_t* __restrict__ dst,
                                 int ktiles, int nstride, int total) {
  for (int tid = blockIdx.x * blockDim.x + threadIdx.x; tid < total;
       tid += gridDim.x * blockDim.x) {
    int v = tid & 7, lane = (tid >> 3) & 31, rem = tid >> 8;
    int kt = rem % ktiles, nt = rem / ktiles;
    int k = kt * 32 + ((v < 4) ? (2 * v + 8 * (lane >> 4)) : (16 + 2 * (v - 4) + 8 * (lane >> 4)));
    int n = nt * 16 + (lane & 15);
    const float* p = src + (size_t)k * nstride + n;
    dst[tid] = pack2(p[0], p[nstride]);
  }
}

// ---- Phase 1: xw = x @ kernel + bias, bf16 C-fragments ---------------------
__global__ __launch_bounds__(256) void xproj_kernel(
    const uint32_t* __restrict__ xswz, const uint32_t* __restrict__ kswz,
    const float* __restrict__ bias, uint4* __restrict__ xw) {
  const int lane = threadIdx.x & 31;
  const int gw = blockIdx.x * 8 + (threadIdx.x >> 5);
  const int mtile = gw >> 6;
  const int n0 = (gw & 63) * 2;
  const float bv0 = bias[n0 * 16 + (lane & 15)];
  const float bv1 = bias[(n0 + 1) * 16 + (lane & 15)];
  f32x8 acc0, acc1;
#pragma unroll
  for (int i = 0; i < 8; i++) { acc0[i] = bv0; acc1[i] = bv1; }
  const uint4* xa = (const uint4*)xswz;
  const uint4* kb = (const uint4*)kswz;
#pragma unroll
  for (int kt = 0; kt < 4; kt++) {
    FragB a, b0, b1;
    int ai = ((mtile * 4 + kt) * 32 + lane) * 2;
    a.q[0] = xa[ai]; a.q[1] = xa[ai + 1];
    int bi0 = ((n0 * 4 + kt) * 32 + lane) * 2;
    int bi1 = (((n0 + 1) * 4 + kt) * 32 + lane) * 2;
    b0.q[0] = kb[bi0]; b0.q[1] = kb[bi0 + 1];
    b1.q[0] = kb[bi1]; b1.q[1] = kb[bi1 + 1];
    acc0 = __builtin_amdgcn_wmma_f32_16x16x32_bf16(false, a.v, false, b0.v, (short)0, acc0, false, false);
    acc1 = __builtin_amdgcn_wmma_f32_16x16x32_bf16(false, a.v, false, b1.v, (short)0, acc1, false, false);
  }
  uint4 o0, o1;
  o0.x = pack2(acc0[0], acc0[1]); o0.y = pack2(acc0[2], acc0[3]);
  o0.z = pack2(acc0[4], acc0[5]); o0.w = pack2(acc0[6], acc0[7]);
  o1.x = pack2(acc1[0], acc1[1]); o1.y = pack2(acc1[2], acc1[3]);
  o1.z = pack2(acc1[4], acc1[5]); o1.w = pack2(acc1[6], acc1[7]);
  size_t oi = ((size_t)mtile * 128 + n0) * 32 + lane;
  xw[oi] = o0;
  xw[oi + 32] = o1;
}

// ---- zero the h exchange buffer + barrier counters (every launch) ----------
__global__ void init_sync_kernel(uint32_t* __restrict__ hbuf, uint32_t* __restrict__ ctr) {
  int tid = blockIdx.x * blockDim.x + threadIdx.x;
  if (tid < 32768) hbuf[tid] = 0;   // 128KB h exchange (both buffers)
  if (tid < 1024)  ctr[tid] = 0;    // 2 groups x 512 step counters
}

// ---- Phase 2: weight-resident recurrence + fused dense ---------------------
// grid = 32: j = blockIdx&15 (h-cols [32j,+32)), grp = blockIdx>>4 (rows [32grp,+32))
// 2 waves/WG: wave s owns cols [32j+16s,+16) and all 4 gate tiles of them.
// hbuf layout: [buf(2)][grp(2)][m(2)][kt(16)][lane(32)][8 dwords]  (A-fragments)
__global__ __launch_bounds__(64, 1) void lstm_kernel(
    const uint4* __restrict__ xw, const uint32_t* __restrict__ rswz,
    const uint32_t* __restrict__ dswz, const float* __restrict__ dense_b,
    uint32_t* __restrict__ hbuf, uint32_t* __restrict__ ctr,
    float* __restrict__ out) {
  const int lane = threadIdx.x & 31;
  const int s = threadIdx.x >> 5;      // wave id = col half
  const int j = blockIdx.x & 15;       // N-split index (= A-fragment ktile it produces)
  const int grp = blockIdx.x >> 4;     // batch group
  const int half = lane >> 4;
  const int nloc = lane & 15;

  // --- rec_kernel slice: 4 gates x 16 ktiles, resident in 512 VGPRs ---
  FragB breg[4][16];
  const uint4* rq = (const uint4*)rswz;
#pragma unroll
  for (int g = 0; g < 4; g++) {
    const int nt = 32 * g + 2 * j + s;
#pragma unroll
    for (int kt = 0; kt < 16; kt++) {
      size_t bi = ((size_t)(nt * 16 + kt) * 32 + lane) * 2;
      breg[g][kt].q[0] = rq[bi];
      breg[g][kt].q[1] = rq[bi + 1];
    }
  }

  f32x8 creg[2], hreg[2];
#pragma unroll
  for (int m = 0; m < 2; m++)
#pragma unroll
    for (int i = 0; i < 8; i++) { creg[m][i] = 0.f; hreg[m][i] = 0.f; }

  const float dbv = (j < 2) ? dense_b[j * 16 + nloc] : 0.f;
  const uint4* hq = (const uint4*)hbuf;
  unsigned short* hss = (unsigned short*)hbuf;
  const uint4* dq = (const uint4*)dswz;

  // h -> A-fragment scatter constants (col = 32j + 16s + nloc, klocal = 16s+nloc)
  const int halfA = (nloc >> 3) & 1;
  const int vA = s * 4 + ((nloc & 7) >> 1);
  const int par = nloc & 1;

  for (int t = 0; t < 512; t++) {
    f32x8 acc[2][4];
    // z init = xw (bias included)
#pragma unroll
    for (int m = 0; m < 2; m++) {
      const int mt = t * 4 + 2 * grp + m;
#pragma unroll
      for (int g = 0; g < 4; g++) {
        const int nt = 32 * g + 2 * j + s;
        uint4 u = xw[((size_t)mt * 128 + nt) * 32 + lane];
        acc[m][g][0] = bflo(u.x); acc[m][g][1] = bfhi(u.x);
        acc[m][g][2] = bflo(u.y); acc[m][g][3] = bfhi(u.y);
        acc[m][g][4] = bflo(u.z); acc[m][g][5] = bfhi(u.z);
        acc[m][g][6] = bflo(u.w); acc[m][g][7] = bfhi(u.w);
      }
    }
    // z += h_{t-1} @ rec_kernel : A from L2-hot exchange buffer, B from VGPRs
    const size_t rbase = (size_t)((t + 1) & 1) * 2 + grp;
#pragma unroll
    for (int m = 0; m < 2; m++) {
#pragma unroll
      for (int kt = 0; kt < 16; kt++) {
        FragB a;
        size_t ai = ((rbase * 2 + m) * 16 + kt) * 32 + lane;
        a.q[0] = hq[ai * 2]; a.q[1] = hq[ai * 2 + 1];
#pragma unroll
        for (int g = 0; g < 4; g++)
          acc[m][g] = __builtin_amdgcn_wmma_f32_16x16x32_bf16(
              false, a.v, false, breg[g][kt].v, (short)0, acc[m][g], false, false);
      }
    }
    // gates in-registers; write h_t bf16 A-fragments (this WG produces ktile j)
    const size_t wbase = (size_t)(t & 1) * 2 + grp;
#pragma unroll
    for (int m = 0; m < 2; m++) {
#pragma unroll
      for (int v = 0; v < 8; v++) {
        float iv = sigmoidf_(acc[m][0][v]);
        float fv = sigmoidf_(acc[m][1][v]);
        float gv = tanhf(acc[m][2][v]);
        float ov = sigmoidf_(acc[m][3][v]);
        float cn = fv * creg[m][v] + iv * gv;
        float hn = ov * tanhf(cn);
        creg[m][v] = cn; hreg[m][v] = hn;
        int laneA = (v + 8 * half) + 16 * halfA;
        size_t sidx = (((wbase * 2 + m) * 16 + j) * 32 + laneA) * 16 + vA * 2 + par;
        hss[sidx] = (unsigned short)bf16rn(hn);
      }
    }
    // ---- group-wide barrier (16 WGs): release stores, count, acquire ----
    __builtin_amdgcn_fence(__ATOMIC_RELEASE, "agent");
    __syncthreads();
    if (threadIdx.x == 0) {
      uint32_t* c = ctr + (grp << 9) + t;
      __hip_atomic_fetch_add(c, 1u, __ATOMIC_RELAXED, __HIP_MEMORY_SCOPE_AGENT);
      while (__hip_atomic_load(c, __ATOMIC_RELAXED, __HIP_MEMORY_SCOPE_AGENT) < 16u)
        __builtin_amdgcn_s_sleep(1);
    }
    __syncthreads();
    __builtin_amdgcn_fence(__ATOMIC_ACQUIRE, "agent");
    // ---- fused Dense(32, tanh): WGs j=0,1; wave s -> rows [32grp+16s,+16) ----
    if (j < 2) {
      f32x8 dacc;
#pragma unroll
      for (int i = 0; i < 8; i++) dacc[i] = dbv;
#pragma unroll
      for (int kt = 0; kt < 16; kt++) {
        FragB a, b;
        size_t ai = ((wbase * 2 + s) * 16 + kt) * 32 + lane;
        a.q[0] = hq[ai * 2]; a.q[1] = hq[ai * 2 + 1];
        size_t bi = ((size_t)(j * 16 + kt) * 32 + lane) * 2;
        b.q[0] = dq[bi]; b.q[1] = dq[bi + 1];
        dacc = __builtin_amdgcn_wmma_f32_16x16x32_bf16(
            false, a.v, false, b.v, (short)0, dacc, false, false);
      }
#pragma unroll
      for (int v = 0; v < 8; v++) {
        int b = 32 * grp + 16 * s + v + 8 * half;
        out[((size_t)b * 512 + t) * 32 + 16 * j + nloc] = tanhf(dacc[v]);
      }
    }
  }
  // final h, c
  float* hout = out + (size_t)64 * 512 * 32;
  float* cout = hout + 64 * 512;
  const int col = 32 * j + 16 * s + nloc;
#pragma unroll
  for (int m = 0; m < 2; m++)
#pragma unroll
    for (int v = 0; v < 8; v++) {
      int b = 32 * grp + 16 * m + v + 8 * half;
      hout[(size_t)b * 512 + col] = hreg[m][v];
      cout[(size_t)b * 512 + col] = creg[m][v];
    }
}

extern "C" void kernel_launch(void* const* d_in, const int* in_sizes, int n_in,
                              void* d_out, int out_size, void* d_ws, size_t ws_size,
                              hipStream_t stream) {
  const float* x       = (const float*)d_in[0];
  const float* kern    = (const float*)d_in[1];
  const float* rec     = (const float*)d_in[2];
  const float* bias    = (const float*)d_in[3];
  const float* dense_w = (const float*)d_in[4];
  const float* dense_b = (const float*)d_in[5];
  float* out = (float*)d_out;

  char* ws = (char*)d_ws;
  uint32_t* xswz = (uint32_t*)(ws);                          // 8 MB
  uint32_t* kswz = (uint32_t*)(ws + ( 8ull << 20));          // 512 KB
  uint32_t* rswz = (uint32_t*)(ws + ( 9ull << 20));          // 2 MB
  uint32_t* dswz = (uint32_t*)(ws + (11ull << 20));          // 32 KB
  uint4*    xw   = (uint4*)   (ws + (12ull << 20));          // 128 MB
  uint32_t* hbuf = (uint32_t*)(ws + (140ull << 20));         // 128 KB
  uint32_t* ctr  = (uint32_t*)(ws + (140ull << 20) + (128ull << 10)); // 4 KB

  // Phase 0: fragment swizzles (bf16 conversion + layout) + sync init
  swizzle_x_kernel<<<8192, 256, 0, stream>>>(x, xswz);
  swizzle_w_kernel<<<512,  256, 0, stream>>>(kern,    kswz,  4, 2048, 128 * 4 * 256);
  swizzle_w_kernel<<<2048, 256, 0, stream>>>(rec,     rswz, 16, 2048, 128 * 16 * 256);
  swizzle_w_kernel<<<32,   256, 0, stream>>>(dense_w, dswz, 16,   32,   2 * 16 * 256);
  init_sync_kernel<<<128, 256, 0, stream>>>(hbuf, ctr);
  // Phase 1: input projection (2048 mtiles x 64 ntile-pairs, 8 waves/block)
  xproj_kernel<<<16384, 256, 0, stream>>>(xswz, kswz, bias, xw);
  // Phase 2: recurrence + fused dense (32 WGs x 2 waves, weights in VGPRs)
  lstm_kernel<<<32, 64, 0, stream>>>(xw, rswz, dswz, dense_b, hbuf, ctr, out);
}
